// AttentionLayer_8091718386135
// MI455X (gfx1250) — matrix operand order repaired and checked by reading, not compiled, not run on hardware
//
#include <hip/hip_runtime.h>

// ---------- types ----------
typedef __attribute__((ext_vector_type(16))) __bf16 bf16x16;
typedef __attribute__((ext_vector_type(8)))  __bf16 bf16x8;
typedef __attribute__((ext_vector_type(8)))  float  f32x8;
typedef __attribute__((ext_vector_type(4)))  int    i32x4;

constexpr int kB = 4;
constexpr int kL = 2048;
constexpr int kA = 1024;   // ATTEN_DIM == Q/K/V dims
constexpr int kH = 16;
constexpr int kD = 64;
constexpr int kM = kB * kL;  // 8192 rows

#define WMMA_BF16(A, B, C) \
    __builtin_amdgcn_wmma_f32_16x16x32_bf16(false, (A), false, (B), (short)0, (C), false, false)

// ---------- gfx1250 async global->LDS copy (ASYNCcnt path), with safe fallback ----------
#define AS1 __attribute__((address_space(1)))
#define AS3 __attribute__((address_space(3)))

#if defined(__HIP_DEVICE_COMPILE__) && defined(__gfx1250__) && \
    __has_builtin(__builtin_amdgcn_global_load_async_to_lds_b128)
#define HAVE_ASYNC_COPY 1
#else
#define HAVE_ASYNC_COPY 0
#endif

__device__ __forceinline__ void async_copy_b128(const __bf16* g, __bf16* l) {
#if HAVE_ASYNC_COPY
    __builtin_amdgcn_global_load_async_to_lds_b128((AS1 i32x4*)g, (AS3 i32x4*)l, 0, 0);
#else
    *reinterpret_cast<bf16x8*>(l) = *reinterpret_cast<const bf16x8*>(g);
#endif
}

__device__ __forceinline__ void wait_async() {
#if HAVE_ASYNC_COPY
#if __has_builtin(__builtin_amdgcn_s_wait_asynccnt)
    __builtin_amdgcn_s_wait_asynccnt(0);
#else
    asm volatile("s_wait_asynccnt 0" ::: "memory");
#endif
#endif
}

// ---------- helpers ----------
__device__ __forceinline__ bf16x8 load_bf8(const __bf16* p) {
    return *reinterpret_cast<const bf16x8*>(p);
}

__device__ __forceinline__ bf16x16 cat8(bf16x8 lo, bf16x8 hi) {
    return __builtin_shufflevector(lo, hi, 0, 1, 2, 3, 4, 5, 6, 7,
                                           8, 9, 10, 11, 12, 13, 14, 15);
}

// load 8 contiguous f32 (two b128 loads) and downconvert to bf16
__device__ __forceinline__ bf16x8 cvt_f32x8(const float* p) {
    const float4* p4 = reinterpret_cast<const float4*>(p);
    float4 f0 = p4[0];
    float4 f1 = p4[1];
    bf16x8 r;
    r[0] = (__bf16)f0.x; r[1] = (__bf16)f0.y; r[2] = (__bf16)f0.z; r[3] = (__bf16)f0.w;
    r[4] = (__bf16)f1.x; r[5] = (__bf16)f1.y; r[6] = (__bf16)f1.z; r[7] = (__bf16)f1.w;
    return r;
}

// ---------- kernel 1: weight convert + transpose  W[K][N] f32 -> Wt[N][K] bf16 ----------
__global__ __launch_bounds__(256)
void wconv_kernel(const float* __restrict__ w, __bf16* __restrict__ wt) {
    int idx = blockIdx.x * blockDim.x + threadIdx.x;  // over 1024*1024, coalesced writes
    int n = idx >> 10;
    int k = idx & 1023;
    wt[idx] = (__bf16)w[(k << 10) + n];
}

// ---------- kernel 2: QKV projection GEMM ----------
// C[M=8192, N=1024] = A_f32[M,K=1024] x Wt_bf16[N,K] + bias
// B tile (64 cols x 32 k) staged per-block via double-buffered async global->LDS.
// vmode 0: out[b,h,pos,d] (Q,K)   vmode 1: out[b,h,d,pos] (V transposed)
__global__ __launch_bounds__(128)
void gemm_x_w_kernel(const float* __restrict__ A, const __bf16* __restrict__ Wt,
                     const float* __restrict__ bias, __bf16* __restrict__ out, int vmode) {
    __shared__ __align__(16) __bf16 ldsB[2][64][40];  // stride 40 bf16 = 80B, bank-conflict-free

    const int lane = threadIdx.x & 31;
    const int wave = threadIdx.x >> 5;
    const int half = lane >> 4;
    const int ln   = lane & 15;
    const int m0 = blockIdx.y * 256 + wave * 64;
    const int n0 = blockIdx.x * 64;

    auto issueB = [&](int buf, int kk) {
        #pragma unroll
        for (int i = 0; i < 2; ++i) {
            int idx = (int)threadIdx.x + 128 * i;   // 0..255 = 64 rows x 4 segs of 16B
            int row = idx >> 2, seg = idx & 3;
            async_copy_b128(Wt + (size_t)(n0 + row) * kA + kk + seg * 8,
                            &ldsB[buf][row][seg * 8]);
        }
    };

    issueB(0, 0);
    f32x8 acc[4][4] = {};
    for (int kk = 0; kk < kA; kk += 32) {
        const int buf = (kk >> 5) & 1;

        // A fragments direct from global (per-wave unique rows), overlap with async copy
        bf16x16 af[4];
        #pragma unroll
        for (int mt = 0; mt < 4; ++mt) {
            // CDNA5 16-bit A layout: lane l holds row (l&15); K chunks at
            // kk + half*8 .. +8 (vgpr0-3) and kk + 16 + half*8 .. +8 (vgpr4-7)
            const float* ap = A + (size_t)(m0 + mt * 16 + ln) * kA + kk + half * 8;
            af[mt] = cat8(cvt_f32x8(ap), cvt_f32x8(ap + 16));
        }

        wait_async();
        __syncthreads();
        if (kk + 32 < kA) issueB(buf ^ 1, kk + 32);

        // B fragments from LDS: lane l holds column (l&15); K = half*8 .. +8 and 16 + half*8 .. +8
        bf16x16 bfr[4];
        #pragma unroll
        for (int nt = 0; nt < 4; ++nt) {
            const __bf16* bp = &ldsB[buf][nt * 16 + ln][half * 8];
            bfr[nt] = cat8(load_bf8(bp), load_bf8(bp + 16));
        }

        #pragma unroll
        for (int mt = 0; mt < 4; ++mt)
            #pragma unroll
            for (int nt = 0; nt < 4; ++nt)
                acc[mt][nt] = WMMA_BF16(af[mt], bfr[nt], acc[mt][nt]);
    }

    float bcol[4];
    #pragma unroll
    for (int nt = 0; nt < 4; ++nt) bcol[nt] = bias[n0 + nt * 16 + ln];

    #pragma unroll
    for (int mt = 0; mt < 4; ++mt) {
        #pragma unroll
        for (int nt = 0; nt < 4; ++nt) {
            #pragma unroll
            for (int r = 0; r < 8; ++r) {
                // C layout: element r of lane l is (m = r + 8*(l>>4), n = l&15)
                int row = m0 + mt * 16 + r + 8 * half;
                int col = n0 + nt * 16 + ln;
                float v = acc[mt][nt][r] + bcol[nt];
                int bb = row >> 11, pos = row & (kL - 1);
                int h = col >> 6, d = col & 63;
                if (vmode == 0)
                    out[(((size_t)bb * kH + h) * kL + pos) * kD + d] = (__bf16)v;
                else
                    out[(((size_t)bb * kH + h) * kD + d) * kL + pos] = (__bf16)v;
            }
        }
    }
}

// ---------- kernel 3: flash attention ----------
// Q, Kr: [b,h,pos,d] bf16 ; Vt: [b,h,d,pos] bf16 ; ctx out: [b*L, A] bf16
// K block (32x64) and V^T block (64x32) are shared by all 4 waves of the block:
// staged once via double-buffered async global->LDS copies.
__global__ __launch_bounds__(128)
void attn_kernel(const __bf16* __restrict__ Q, const __bf16* __restrict__ Kr,
                 const __bf16* __restrict__ Vt, __bf16* __restrict__ ctx) {
    __shared__ __align__(16) __bf16 ldsK[2][32][72];  // 32 keys x 64 d, stride 72 (144B)
    __shared__ __align__(16) __bf16 ldsV[2][64][40];  // 64 d x 32 pos, stride 40 (80B)
    __shared__ __align__(16) __bf16 ldsP[4][16][40];  // per-wave P tile (16 q x 32 keys)

    const int lane = threadIdx.x & 31;
    const int wave = threadIdx.x >> 5;
    const int half = lane >> 4;
    const int ln   = lane & 15;
    const int b = blockIdx.z, h = blockIdx.y;
    const int q0 = blockIdx.x * 64 + wave * 16;
    const int bh = b * kH + h;

    const __bf16* qb  = Q  + (size_t)bh * kL * kD;
    const __bf16* kbp = Kr + (size_t)bh * kL * kD;
    const __bf16* vb  = Vt + (size_t)bh * kD * kL;

    auto issueKV = [&](int buf, int kb0) {
        #pragma unroll
        for (int i = 0; i < 2; ++i) {
            int idx = (int)threadIdx.x + 128 * i;       // 0..255
            int krow = idx >> 3, kseg = idx & 7;        // 32 rows x 8 segs (128B rows)
            async_copy_b128(kbp + (size_t)(kb0 + krow) * kD + kseg * 8,
                            &ldsK[buf][krow][kseg * 8]);
            int vrow = idx >> 2, vseg = idx & 3;        // 64 rows x 4 segs (64B rows)
            async_copy_b128(vb + (size_t)vrow * kL + kb0 + vseg * 8,
                            &ldsV[buf][vrow][vseg * 8]);
        }
    };

    // Q fragments (A layout), 16 rows x 64 d, loaded once
    bf16x16 aq[2];
    #pragma unroll
    for (int ks = 0; ks < 2; ++ks) {
        const __bf16* qp = qb + (size_t)(q0 + ln) * kD + ks * 32 + half * 8;
        aq[ks] = cat8(load_bf8(qp), load_bf8(qp + 16));
    }

    issueKV(0, 0);

    f32x8 acc[4] = {};
    float mprev[8], lsum[8];
    #pragma unroll
    for (int r = 0; r < 8; ++r) { mprev[r] = -1e30f; lsum[r] = 0.0f; }

    for (int kb0 = 0; kb0 < kL; kb0 += 32) {
        const int buf = (kb0 >> 5) & 1;

        wait_async();
        __syncthreads();
        if (kb0 + 32 < kL) issueKV(buf ^ 1, kb0 + 32);
        if (kb0 + 64 < kL)
            __builtin_prefetch(kbp + (size_t)(kb0 + 64 + ln) * kD, 0, 3);

        // ---- scores S = Q x K^T for 32 keys (two 16-key tiles), K from LDS ----
        f32x8 c0 = {}, c1 = {};
        #pragma unroll
        for (int ks = 0; ks < 2; ++ks) {
            const __bf16* kp0 = &ldsK[buf][ln][ks * 32 + half * 8];
            const __bf16* kp1 = &ldsK[buf][16 + ln][ks * 32 + half * 8];
            bf16x16 bk0 = cat8(load_bf8(kp0), load_bf8(kp0 + 16));
            bf16x16 bk1 = cat8(load_bf8(kp1), load_bf8(kp1 + 16));
            c0 = WMMA_BF16(aq[ks], bk0, c0);
            c1 = WMMA_BF16(aq[ks], bk1, c1);
        }

        // ---- online softmax (reductions across the 16-lane half owning each row) ----
        float corr[8];
        #pragma unroll
        for (int r = 0; r < 8; ++r) {
            float s0 = c0[r] * 0.125f;   // 1/sqrt(64)
            float s1 = c1[r] * 0.125f;
            float mx = fmaxf(s0, s1);
            #pragma unroll
            for (int off = 8; off >= 1; off >>= 1)
                mx = fmaxf(mx, __shfl_xor(mx, off, 32));
            float nm = fmaxf(mprev[r], mx);
            float p0 = __expf(s0 - nm);
            float p1 = __expf(s1 - nm);
            float rs = p0 + p1;
            #pragma unroll
            for (int off = 8; off >= 1; off >>= 1)
                rs += __shfl_xor(rs, off, 32);
            corr[r] = __expf(mprev[r] - nm);
            lsum[r] = lsum[r] * corr[r] + rs;
            mprev[r] = nm;
            int m = r + 8 * half;
            ldsP[wave][m][ln]      = (__bf16)p0;
            ldsP[wave][m][16 + ln] = (__bf16)p1;
        }
        #pragma unroll
        for (int dt = 0; dt < 4; ++dt)
            #pragma unroll
            for (int r = 0; r < 8; ++r)
                acc[dt][r] *= corr[r];

        // LDS round-trip: C-layout -> A-layout (per-wave private tile, DS in-order)
        asm volatile("s_wait_dscnt 0" ::: "memory");
        const __bf16* pp = &ldsP[wave][ln][half * 8];
        bf16x16 apf = cat8(*(const bf16x8*)pp, *(const bf16x8*)(pp + 16));

        // ---- O += P x V, V from LDS ----
        #pragma unroll
        for (int dt = 0; dt < 4; ++dt) {
            const __bf16* vp = &ldsV[buf][dt * 16 + ln][half * 8];
            bf16x16 bv = cat8(load_bf8(vp), load_bf8(vp + 16));
            acc[dt] = WMMA_BF16(apf, bv, acc[dt]);
        }
    }

    // normalize and write context [b*L, A]
    #pragma unroll
    for (int dt = 0; dt < 4; ++dt) {
        #pragma unroll
        for (int r = 0; r < 8; ++r) {
            int m = r + 8 * half;
            float v = acc[dt][r] / lsum[r];
            int qrow = q0 + m;
            int d = dt * 16 + ln;
            ctx[((size_t)b * kL + qrow) * kA + h * kD + d] = (__bf16)v;
        }
    }
}

// ---------- kernel 4: output projection GEMM (bf16 A, f32 out) ----------
__global__ __launch_bounds__(128)
void gemm_ctx_wo_kernel(const __bf16* __restrict__ A, const __bf16* __restrict__ Wt,
                        const float* __restrict__ bias, float* __restrict__ out) {
    __shared__ __align__(16) __bf16 ldsB[2][64][40];

    const int lane = threadIdx.x & 31;
    const int wave = threadIdx.x >> 5;
    const int half = lane >> 4;
    const int ln   = lane & 15;
    const int m0 = blockIdx.y * 256 + wave * 64;
    const int n0 = blockIdx.x * 64;

    auto issueB = [&](int buf, int kk) {
        #pragma unroll
        for (int i = 0; i < 2; ++i) {
            int idx = (int)threadIdx.x + 128 * i;
            int row = idx >> 2, seg = idx & 3;
            async_copy_b128(Wt + (size_t)(n0 + row) * kA + kk + seg * 8,
                            &ldsB[buf][row][seg * 8]);
        }
    };

    issueB(0, 0);
    f32x8 acc[4][4] = {};
    for (int kk = 0; kk < kA; kk += 32) {
        const int buf = (kk >> 5) & 1;

        bf16x16 af[4];
        #pragma unroll
        for (int mt = 0; mt < 4; ++mt) {
            const __bf16* ap = A + (size_t)(m0 + mt * 16 + ln) * kA + kk + half * 8;
            af[mt] = cat8(load_bf8(ap), load_bf8(ap + 16));
        }

        wait_async();
        __syncthreads();
        if (kk + 32 < kA) issueB(buf ^ 1, kk + 32);

        bf16x16 bfr[4];
        #pragma unroll
        for (int nt = 0; nt < 4; ++nt) {
            const __bf16* bp = &ldsB[buf][nt * 16 + ln][half * 8];
            bfr[nt] = cat8(load_bf8(bp), load_bf8(bp + 16));
        }

        #pragma unroll
        for (int mt = 0; mt < 4; ++mt)
            #pragma unroll
            for (int nt = 0; nt < 4; ++nt)
                acc[mt][nt] = WMMA_BF16(af[mt], bfr[nt], acc[mt][nt]);
    }

    float bcol[4];
    #pragma unroll
    for (int nt = 0; nt < 4; ++nt) bcol[nt] = bias[n0 + nt * 16 + ln];

    #pragma unroll
    for (int mt = 0; mt < 4; ++mt)
        #pragma unroll
        for (int nt = 0; nt < 4; ++nt)
            #pragma unroll
            for (int r = 0; r < 8; ++r) {
                int row = m0 + mt * 16 + r + 8 * half;
                int col = n0 + nt * 16 + ln;
                out[(size_t)row * kA + col] = acc[mt][nt][r] + bcol[nt];
            }
}

// ---------- host launcher ----------
extern "C" void kernel_launch(void* const* d_in, const int* in_sizes, int n_in,
                              void* d_out, int out_size, void* d_ws, size_t ws_size,
                              hipStream_t stream) {
    (void)in_sizes; (void)n_in; (void)out_size; (void)ws_size;

    const float* q  = (const float*)d_in[0];
    const float* k  = (const float*)d_in[1];
    const float* v  = (const float*)d_in[2];
    const float* Wq = (const float*)d_in[3];
    const float* bq = (const float*)d_in[4];
    const float* Wk = (const float*)d_in[5];
    const float* bk = (const float*)d_in[6];
    const float* Wv = (const float*)d_in[7];
    const float* bv = (const float*)d_in[8];
    const float* Wo = (const float*)d_in[9];
    const float* bo = (const float*)d_in[10];
    float* out = (float*)d_out;

    // workspace layout (bf16 elements): 4x weightT (1M each) + Q + K + Vt + ctx (8M each) = 72 MB
    __bf16* wsb = (__bf16*)d_ws;
    const size_t WSZ = (size_t)kA * kA;
    const size_t XSZ = (size_t)kM * kA;
    __bf16* wq_t = wsb;
    __bf16* wk_t = wq_t + WSZ;
    __bf16* wv_t = wk_t + WSZ;
    __bf16* wo_t = wv_t + WSZ;
    __bf16* qw   = wo_t + WSZ;
    __bf16* kw   = qw + XSZ;
    __bf16* vt   = kw + XSZ;
    __bf16* ctx  = vt + XSZ;

    dim3 blk(128);
    int cgrid = (kA * kA) / 256;
    wconv_kernel<<<cgrid, 256, 0, stream>>>(Wq, wq_t);
    wconv_kernel<<<cgrid, 256, 0, stream>>>(Wk, wk_t);
    wconv_kernel<<<cgrid, 256, 0, stream>>>(Wv, wv_t);
    wconv_kernel<<<cgrid, 256, 0, stream>>>(Wo, wo_t);

    dim3 gg(kA / 64, kM / 256);  // (16, 32)
    gemm_x_w_kernel<<<gg, blk, 0, stream>>>(q, wq_t, bq, qw, 0);
    gemm_x_w_kernel<<<gg, blk, 0, stream>>>(k, wk_t, bk, kw, 0);
    gemm_x_w_kernel<<<gg, blk, 0, stream>>>(v, wv_t, bv, vt, 1);

    dim3 ga(kL / 64, kH, kB);    // (32, 16, 4)
    attn_kernel<<<ga, blk, 0, stream>>>(qw, kw, vt, ctx);

    gemm_ctx_wo_kernel<<<gg, blk, 0, stream>>>(ctx, wo_t, bo, out);
}
